// GAT_66005057405149
// MI455X (gfx1250) — matrix operand order repaired and checked by reading, not compiled, not run on hardware
//
#include <hip/hip_runtime.h>
#include <math.h>

typedef __attribute__((ext_vector_type(16))) _Float16 v16h;
typedef __attribute__((ext_vector_type(8)))  float    v8f;

#define NN   100000
#define FIN  128
#define DD   64
#define HH   8
#define GG   64
#define NCLS 10
#define SLOPE 0.2f

// ---------------------------------------------------------------------------
// Projection GEMM: Hout[N,64] = A[N,Din] @ W[Din,64], via v_wmma_f32_16x16x32_f16
// block = 128 threads (4 waves). Wave w computes the 16x16 tile covering
// columns [16w, 16w+16); block covers 16 rows. EXEC stays all-ones through WMMA.
// A-fragment loads are b128-vectorized; store guard is a uniform scalar branch.
// ---------------------------------------------------------------------------
__global__ void gat_gemm_wmma(const float* __restrict__ A,
                              const float* __restrict__ W,
                              float* __restrict__ Hout,
                              int n, int Din) {
  const int wave = threadIdx.x >> 5;          // column tile 0..3
  const int lane = threadIdx.x & 31;
  const int l16  = lane & 15;
  const int hi   = lane >> 4;                 // lane-half select
  const int rowBase = blockIdx.x * 16;

  // branchless row clamp so WMMA runs with full EXEC; stores handled below
  int rowA = rowBase + l16;
  rowA = rowA < n ? rowA : n - 1;

  const int col = wave * 16 + l16;
  const int khalf = hi * 8;                   // A K-offset within 16-wide half
  const int kb    = hi * 16;                  // unused; B loads use khalf like A

  v8f c = {};
  for (int k0 = 0; k0 < Din; k0 += 32) {
    // A fragment: lanes 0-15 row M=l16 K=[0..7,16..23]; lanes 16-31 K=[8..15,24..31]
    const float* arow = A + (size_t)rowA * Din + k0 + khalf;   // 32B-aligned
    const float4 a0 = ((const float4*)arow)[0];                // K 0..3   (rel)
    const float4 a1 = ((const float4*)arow)[1];                // K 4..7
    const float4 a2 = ((const float4*)(arow + 16))[0];         // K 16..19
    const float4 a3 = ((const float4*)(arow + 16))[1];         // K 20..23
    v16h a, b;
    a[0]  = (_Float16)a0.x; a[1]  = (_Float16)a0.y; a[2]  = (_Float16)a0.z; a[3]  = (_Float16)a0.w;
    a[4]  = (_Float16)a1.x; a[5]  = (_Float16)a1.y; a[6]  = (_Float16)a1.z; a[7]  = (_Float16)a1.w;
    a[8]  = (_Float16)a2.x; a[9]  = (_Float16)a2.y; a[10] = (_Float16)a2.z; a[11] = (_Float16)a2.w;
    a[12] = (_Float16)a3.x; a[13] = (_Float16)a3.y; a[14] = (_Float16)a3.z; a[15] = (_Float16)a3.w;
    // B fragment: lanes 0-15 hold K=[0..7,16..23], lanes 16-31 K=[8..15,24..31] (rel k0), col N=l16
#pragma unroll
    for (int i = 0; i < 16; ++i) b[i] = (_Float16)W[(size_t)(k0 + khalf + i + (i >= 8 ? 8 : 0)) * DD + col];
    c = __builtin_amdgcn_wmma_f32_16x16x32_f16(false, a, false, b, (short)0, c, false, false);
  }
  // D layout: VGPR j, lanes 0-15 -> M=j, lanes 16-31 -> M=8+j, N = lane&15
  if (rowBase + 16 <= n) {            // uniform branch: full tile, EXEC stays all-ones
#pragma unroll
    for (int j = 0; j < 8; ++j)
      Hout[(size_t)(rowBase + hi * 8 + j) * DD + col] = c[j];
  } else {
#pragma unroll
    for (int j = 0; j < 8; ++j) {
      int row = rowBase + hi * 8 + j;
      if (row < n) Hout[(size_t)row * DD + col] = c[j];
    }
  }
}

// ---------------------------------------------------------------------------
// init per layer: accum = 0, m = -inf, denom = 0
// ---------------------------------------------------------------------------
__global__ void gat_init(float* __restrict__ acc, float* __restrict__ m,
                         float* __restrict__ den, int n) {
  int t = blockIdx.x * blockDim.x + threadIdx.x;
  if (t < n * DD) acc[t] = 0.0f;
  if (t < n * HH) { m[t] = -__builtin_inff(); den[t] = 0.0f; }
}

// ---------------------------------------------------------------------------
// attention logits per node/head: als[n,h] = sum_c h[n,h,c]*a_src[h,c]
// ---------------------------------------------------------------------------
__global__ void gat_alpha(const float* __restrict__ H,
                          const float* __restrict__ a_src,
                          const float* __restrict__ a_dst,
                          float* __restrict__ als, float* __restrict__ ald, int n) {
  int t = blockIdx.x * blockDim.x + threadIdx.x;
  if (t >= n * HH) return;
  int node = t >> 3, h = t & 7;
  const float* hp = H + (size_t)node * DD + h * 8;
  const float4 h0 = ((const float4*)hp)[0];
  const float4 h1 = ((const float4*)hp)[1];
  const float4 s0 = ((const float4*)(a_src + h * 8))[0];
  const float4 s1 = ((const float4*)(a_src + h * 8))[1];
  const float4 d0 = ((const float4*)(a_dst + h * 8))[0];
  const float4 d1 = ((const float4*)(a_dst + h * 8))[1];
  float s = h0.x*s0.x + h0.y*s0.y + h0.z*s0.z + h0.w*s0.w
          + h1.x*s1.x + h1.y*s1.y + h1.z*s1.z + h1.w*s1.w;
  float d = h0.x*d0.x + h0.y*d0.y + h0.z*d0.z + h0.w*d0.w
          + h1.x*d1.x + h1.y*d1.y + h1.z*d1.z + h1.w*d1.w;
  als[t] = s; ald[t] = d;
}

__device__ __forceinline__ void atomic_max_f32(float* addr, float v) {
  // sign-aware bit trick: valid when all writers use it and init is -inf
  if (v >= 0.0f) atomicMax((int*)addr, __float_as_int(v));
  else           atomicMin((unsigned int*)addr, __float_as_uint(v));
}

// ---------------------------------------------------------------------------
// edge pass 1: segment max of leakyrelu(als[src]+ald[dst]) over dst
// one thread per (edge, head); edges >= E are self-loops
// ---------------------------------------------------------------------------
__global__ void gat_edge_max(const int* __restrict__ src, const int* __restrict__ dst,
                             const float* __restrict__ als, const float* __restrict__ ald,
                             float* __restrict__ m, int Ne, int n) {
  int t = blockIdx.x * blockDim.x + threadIdx.x;
  int tot = (Ne + n) * HH;
  if (t >= tot) return;
  int e = t >> 3, h = t & 7;
  int s, d;
  if (e < Ne) { s = src[e]; d = dst[e]; } else { s = d = e - Ne; }
  float v = als[s * HH + h] + ald[d * HH + h];
  v = v > 0.0f ? v : SLOPE * v;
  atomic_max_f32(&m[d * HH + h], v);
}

// ---------------------------------------------------------------------------
// edge pass 2: w = exp(e - m[dst]); denom[dst,h] += w; accum[dst,h,:] += w*h[src,h,:]
// h[src] gather is b128-vectorized (dominant traffic, L2-resident)
// ---------------------------------------------------------------------------
__global__ void gat_edge_acc(const int* __restrict__ src, const int* __restrict__ dst,
                             const float* __restrict__ als, const float* __restrict__ ald,
                             const float* __restrict__ m, const float* __restrict__ H,
                             float* __restrict__ den, float* __restrict__ acc,
                             int Ne, int n) {
  int t = blockIdx.x * blockDim.x + threadIdx.x;
  int tot = (Ne + n) * HH;
  if (t >= tot) return;
  int e = t >> 3, h = t & 7;
  int s, d;
  if (e < Ne) { s = src[e]; d = dst[e]; } else { s = d = e - Ne; }
  float v = als[s * HH + h] + ald[d * HH + h];
  v = v > 0.0f ? v : SLOPE * v;
  float w = __expf(v - m[d * HH + h]);
  atomicAdd(&den[d * HH + h], w);
  const float* hp = H + (size_t)s * DD + h * 8;
  const float4 h0 = ((const float4*)hp)[0];
  const float4 h1 = ((const float4*)hp)[1];
  float* ap = acc + (size_t)d * DD + h * 8;
  atomicAdd(&ap[0], w * h0.x); atomicAdd(&ap[1], w * h0.y);
  atomicAdd(&ap[2], w * h0.z); atomicAdd(&ap[3], w * h0.w);
  atomicAdd(&ap[4], w * h1.x); atomicAdd(&ap[5], w * h1.y);
  atomicAdd(&ap[6], w * h1.z); atomicAdd(&ap[7], w * h1.w);
}

// ---------------------------------------------------------------------------
// finalize node: acc = elu(acc/denom + bias)   (in place; acc becomes next input)
// ---------------------------------------------------------------------------
__global__ void gat_node_fin(float* __restrict__ acc, const float* __restrict__ den,
                             const float* __restrict__ bias, int n) {
  int t = blockIdx.x * blockDim.x + threadIdx.x;
  if (t >= n * DD) return;
  int node = t >> 6, d = t & 63, h = d >> 3;
  float v = acc[t] / den[node * HH + h] + bias[d];
  acc[t] = v > 0.0f ? v : (__expf(v) - 1.0f);
}

// ---------------------------------------------------------------------------
// pooling
// ---------------------------------------------------------------------------
__global__ void pool_init(float* __restrict__ sums, float* __restrict__ cnts) {
  int t = blockIdx.x * blockDim.x + threadIdx.x;
  if (t < GG * DD) sums[t] = 0.0f;
  if (t < GG) cnts[t] = 0.0f;
}

__global__ void pool_accum(const float* __restrict__ H, const int* __restrict__ batch,
                           float* __restrict__ sums, float* __restrict__ cnts, int n) {
  int t = blockIdx.x * blockDim.x + threadIdx.x;
  if (t >= n * DD) return;
  int node = t >> 6, d = t & 63;
  int g = batch[node];
  atomicAdd(&sums[g * DD + d], H[t]);
  if (d == 0) atomicAdd(&cnts[g], 1.0f);
}

__global__ void head_kernel(const float* __restrict__ sums, const float* __restrict__ cnts,
                            const float* __restrict__ linW, const float* __restrict__ linb,
                            float* __restrict__ out) {
  int t = threadIdx.x;  // GG*NCLS = 640
  if (t >= GG * NCLS) return;
  int g = t / NCLS, cls = t % NCLS;
  float inv = 1.0f / fmaxf(cnts[g], 1.0f);
  float accv = linb[cls];
  for (int d = 0; d < DD; ++d) accv += sums[g * DD + d] * inv * linW[d * NCLS + cls];
  out[t] = accv;
}

// ---------------------------------------------------------------------------
extern "C" void kernel_launch(void* const* d_in, const int* in_sizes, int n_in,
                              void* d_out, int out_size, void* d_ws, size_t ws_size,
                              hipStream_t stream) {
  const float* x          = (const float*)d_in[0];
  const int*   edge_index = (const int*)  d_in[1];
  const int*   batch      = (const int*)  d_in[2];
  const float* Wl[3]   = { (const float*)d_in[3],  (const float*)d_in[7],  (const float*)d_in[11] };
  const float* asrc[3] = { (const float*)d_in[4],  (const float*)d_in[8],  (const float*)d_in[12] };
  const float* adst[3] = { (const float*)d_in[5],  (const float*)d_in[9],  (const float*)d_in[13] };
  const float* bl[3]   = { (const float*)d_in[6],  (const float*)d_in[10], (const float*)d_in[14] };
  const float* linW = (const float*)d_in[15];
  const float* linb = (const float*)d_in[16];
  float* out = (float*)d_out;

  const int n  = in_sizes[0] / FIN;    // 100000
  const int Ne = in_sizes[1] / 2;      // 1600000
  const int* src = edge_index;
  const int* dst = edge_index + Ne;

  // workspace layout (floats), ~64 MB total
  float* P    = (float*)d_ws;          // [n*64] projected features
  float* ACC  = P    + (size_t)n * DD; // [n*64] attention accumulator / layer output
  float* als  = ACC  + (size_t)n * DD; // [n*8]
  float* ald  = als  + (size_t)n * HH; // [n*8]
  float* m    = ald  + (size_t)n * HH; // [n*8]
  float* den  = m    + (size_t)n * HH; // [n*8]
  float* sums = den  + (size_t)n * HH; // [64*64]
  float* cnts = sums + GG * DD;        // [64]

  const int TB = 256;
  const int gNode64 = (n * DD + TB - 1) / TB;
  const int gNode8  = (n * HH + TB - 1) / TB;
  const int gEdge   = ((Ne + n) * HH + TB - 1) / TB;
  const int gRows   = (n + 15) / 16;

  const float* cur = x;
  for (int l = 0; l < 3; ++l) {
    const int Din = (l == 0) ? FIN : DD;
    gat_gemm_wmma<<<gRows, 128, 0, stream>>>(cur, Wl[l], P, n, Din);
    gat_init<<<gNode64, TB, 0, stream>>>(ACC, m, den, n);
    gat_alpha<<<gNode8, TB, 0, stream>>>(P, asrc[l], adst[l], als, ald, n);
    gat_edge_max<<<gEdge, TB, 0, stream>>>(src, dst, als, ald, m, Ne, n);
    gat_edge_acc<<<gEdge, TB, 0, stream>>>(src, dst, als, ald, m, P, den, ACC, Ne, n);
    gat_node_fin<<<gNode64, TB, 0, stream>>>(ACC, den, bl[l], n);
    cur = ACC;   // finalized layer output feeds next projection
  }

  pool_init<<<(GG * DD + TB - 1) / TB, TB, 0, stream>>>(sums, cnts);
  pool_accum<<<gNode64, TB, 0, stream>>>(ACC, batch, sums, cnts, n);
  head_kernel<<<1, 640, 0, stream>>>(sums, cnts, linW, linb, out);
}
